// MultiHeadAttention_31026843747219
// MI455X (gfx1250) — matrix operand order repaired and checked by reading, not compiled, not run on hardware
//
#include <hip/hip_runtime.h>

// ---------------------------------------------------------------------------
// MultiHeadAttention forward for MI455X (gfx1250, wave32, WMMA)
// B=2, S=4096, D=512, H=8, HD=64
//   1) Q/K/V projections: bf16 WMMA tiled GEMM -> bf16 [B,H,S,HD]
//   2) fused flash attention, transposed-score formulation:
//        S^T = K x Q^T   (C-layout of S^T == A-layout of P, no LDS round trip)
//        O   = P x V     (V consumed in natural [key][hd] layout)
//      K/V staged to LDS with async b128 loads, double buffered.
//   3) O projection: bf16 WMMA GEMM -> fp32 d_out
// ---------------------------------------------------------------------------

typedef __bf16 bf16;
typedef bf16  bf16x4 __attribute__((ext_vector_type(4)));
typedef bf16  bf16x8 __attribute__((ext_vector_type(8)));
typedef bf16  v16bf  __attribute__((ext_vector_type(16)));
typedef float v8f    __attribute__((ext_vector_type(8)));

union V16U { v16bf v; bf16x8 h[2]; };

static __device__ inline v16bf make_v16(bf16x8 lo, bf16x8 hi) {
    V16U u;
    u.h[0] = lo;
    u.h[1] = hi;
    return u.v;
}

constexpr int Bc  = 2;
constexpr int Sq  = 4096;
constexpr int Dm  = 512;
constexpr int Hn  = 8;
constexpr int HDm = 64;
constexpr int KBk = 64;            // keys per block in attention
constexpr int NBk = Sq / KBk;      // 64 key blocks

// --------- CDNA5 async global->LDS staging (guarded, sync fallback) --------
#if __has_builtin(__builtin_amdgcn_global_load_async_to_lds_b128) && \
    __has_builtin(__builtin_amdgcn_s_wait_asynccnt)
#define ATTN_ASYNC 1
typedef int v4i_vs __attribute__((vector_size(16)));   // int4, matches builtin
typedef __attribute__((address_space(1))) v4i_vs gv4i; // global (AS1)
typedef __attribute__((address_space(3))) v4i_vs lv4i; // LDS (AS3)
static __device__ inline gv4i* as_global(const void* p) {
    return (gv4i*)(uintptr_t)p;
}
static __device__ inline lv4i* as_lds(void* p) {
    // generic LDS pointers carry the LDS offset in the low 32 bits
    return (lv4i*)(uint32_t)(uintptr_t)p;
}
#else
#define ATTN_ASYNC 0
#endif

// ---------------------------------------------------------------------------
// Tiled GEMM: Out[m,n] = sum_k X[m,k] * W[n,k] + bias[n]   (torch Linear)
// 256 threads (8 waves), tile 32(m) x 64(n), K-step 32, one
// v_wmma_f32_16x16x32_bf16 per wave per K-step.
// ---------------------------------------------------------------------------
template <bool IN_BF16, bool OUT_HEADS>
__global__ __launch_bounds__(256) void proj_kernel(const void* __restrict__ Xv,
                                                   const float* __restrict__ W,
                                                   const float* __restrict__ bias,
                                                   void* __restrict__ Out) {
    const int m0   = blockIdx.x * 32;
    const int n0   = blockIdx.y * 64;
    const int tid  = threadIdx.x;
    const int lane = tid & 31;
    const int wave = tid >> 5;
    const int wr   = wave & 1;
    const int wc   = wave >> 1;

    __shared__ alignas(16) bf16 Xs[32][32];  // [m][k]
    __shared__ alignas(16) bf16 Wt[32][64];  // [k][n]

    v8f c = {};
    const int half = lane >> 4;
    const int lr   = lane & 15;
    const int kh   = half * 8;

    for (int k0 = 0; k0 < Dm; k0 += 32) {
        __syncthreads();
        {   // stage X tile (convert to bf16)
            const int row = tid >> 3;
            const int kc  = (tid & 7) * 4;
            if constexpr (IN_BF16) {
                const bf16* X = (const bf16*)Xv;
                bf16x4 v = *(const bf16x4*)&X[(size_t)(m0 + row) * Dm + k0 + kc];
#pragma unroll
                for (int j = 0; j < 4; ++j) Xs[row][kc + j] = v[j];
            } else {
                const float* X = (const float*)Xv;
                float4 v = *(const float4*)&X[(size_t)(m0 + row) * Dm + k0 + kc];
                Xs[row][kc + 0] = (bf16)v.x;
                Xs[row][kc + 1] = (bf16)v.y;
                Xs[row][kc + 2] = (bf16)v.z;
                Xs[row][kc + 3] = (bf16)v.w;
            }
        }
        {   // stage W tile transposed: Wt[k][n]
            const int n  = tid >> 2;
            const int kc = (tid & 3) * 8;
            const float* Wr = &W[(size_t)(n0 + n) * Dm + k0 + kc];
            float4 a = *(const float4*)Wr;
            float4 b = *(const float4*)(Wr + 4);
            Wt[kc + 0][n] = (bf16)a.x;
            Wt[kc + 1][n] = (bf16)a.y;
            Wt[kc + 2][n] = (bf16)a.z;
            Wt[kc + 3][n] = (bf16)a.w;
            Wt[kc + 4][n] = (bf16)b.x;
            Wt[kc + 5][n] = (bf16)b.y;
            Wt[kc + 6][n] = (bf16)b.z;
            Wt[kc + 7][n] = (bf16)b.w;
        }
        __syncthreads();

        bf16x8 alo = *(const bf16x8*)&Xs[wr * 16 + lr][kh];
        bf16x8 ahi = *(const bf16x8*)&Xs[wr * 16 + lr][kh + 16];
        v16bf a = make_v16(alo, ahi);
        bf16x8 blo, bhi;
#pragma unroll
        for (int j = 0; j < 8; ++j) {
            blo[j] = Wt[kh + j][wc * 16 + lr];
            bhi[j] = Wt[kh + 16 + j][wc * 16 + lr];
        }
        v16bf b = make_v16(blo, bhi);

        c = __builtin_amdgcn_wmma_f32_16x16x32_bf16(false, a, false, b,
                                                    (short)0, c, false, false);
    }

    const int n  = n0 + wc * 16 + lr;
    const float bv = bias[n];
#pragma unroll
    for (int r = 0; r < 8; ++r) {
        const int m   = m0 + wr * 16 + r + half * 8;
        const float v = c[r] + bv;
        if constexpr (OUT_HEADS) {
            const int bb = m / Sq, s = m % Sq;
            const int h = n / HDm, hd = n % HDm;
            ((bf16*)Out)[(((size_t)bb * Hn + h) * Sq + s) * HDm + hd] = (bf16)v;
        } else {
            ((float*)Out)[(size_t)m * Dm + n] = v;
        }
    }
}

// ---------------------------------------------------------------------------
// Fused flash attention (transposed-score formulation).
// Grid: (S/128, B*H). Block: 256 threads = 8 waves; each wave owns 16 queries.
// ---------------------------------------------------------------------------
__global__ __launch_bounds__(256) void attn_kernel(const bf16* __restrict__ Q,
                                                   const bf16* __restrict__ K,
                                                   const bf16* __restrict__ V,
                                                   const int* __restrict__ mask,
                                                   bf16* __restrict__ Outb) {
    const int qblk = blockIdx.x;
    const int bh   = blockIdx.y;
    const int b    = bh / Hn;
    const int h    = bh % Hn;
    const int tid  = threadIdx.x;
    const int lane = tid & 31;
    const int wave = tid >> 5;
    const int q0   = qblk * 128 + wave * 16;
    const size_t baseH = (size_t)bh * Sq * HDm;

    __shared__ alignas(16) bf16 Ks[2][KBk][HDm];   // [key][hd], A operand of K*Q^T
    __shared__ alignas(16) bf16 Vs[2][KBk][HDm];   // [key][hd], B operand of P*V
    __shared__ alignas(16) bf16 Qt[8][HDm][16];    // per-wave Q^T (B operand)

    const int half = lane >> 4;
    const int lr   = lane & 15;
    const int kh8  = half * 8;

    // ---- build Q^T once: lane covers query lr, hd half*32..+31 ----
    {
        const bf16* qbase = &Q[baseH + (size_t)(q0 + lr) * HDm + half * 32];
#pragma unroll
        for (int c2 = 0; c2 < 4; ++c2) {
            bf16x8 qv = *(const bf16x8*)(qbase + c2 * 8);
#pragma unroll
            for (int j = 0; j < 8; ++j)
                Qt[wave][half * 32 + c2 * 8 + j][lr] = qv[j];
        }
    }
    v16bf qb[2];   // B operands: lane = query column, 16 hd values packed
#pragma unroll
    for (int s2 = 0; s2 < 2; ++s2) {
        bf16x8 lo = *(const bf16x8*)&Q[baseH + (size_t)(q0 + lr) * HDm + s2 * 32 + kh8];
        bf16x8 hi = *(const bf16x8*)&Q[baseH + (size_t)(q0 + lr) * HDm + s2 * 32 + kh8 + 16];
        qb[s2] = make_v16(lo, hi);
    }

    // ---- K/V staging: 256 threads x 2x b128 per tensor per block ----
    const int srow = tid >> 2;         // 0..63 key row
    const int shc  = (tid & 3) * 16;   // hd chunk 0/16/32/48
    auto stage = [&](int kb, int buf) {
        const size_t goff = baseH + (size_t)(kb * KBk + srow) * HDm + shc;
#if ATTN_ASYNC
        __builtin_amdgcn_global_load_async_to_lds_b128(
            as_global(&K[goff]), as_lds(&Ks[buf][srow][shc]), 0, 0);
        __builtin_amdgcn_global_load_async_to_lds_b128(
            as_global(&K[goff + 8]), as_lds(&Ks[buf][srow][shc + 8]), 0, 0);
        __builtin_amdgcn_global_load_async_to_lds_b128(
            as_global(&V[goff]), as_lds(&Vs[buf][srow][shc]), 0, 0);
        __builtin_amdgcn_global_load_async_to_lds_b128(
            as_global(&V[goff + 8]), as_lds(&Vs[buf][srow][shc + 8]), 0, 0);
#else
        *(bf16x8*)&Ks[buf][srow][shc]     = *(const bf16x8*)&K[goff];
        *(bf16x8*)&Ks[buf][srow][shc + 8] = *(const bf16x8*)&K[goff + 8];
        *(bf16x8*)&Vs[buf][srow][shc]     = *(const bf16x8*)&V[goff];
        *(bf16x8*)&Vs[buf][srow][shc + 8] = *(const bf16x8*)&V[goff + 8];
#endif
    };

    v8f acc[4] = {};                   // O tile, C layout: lane = hd col, rows = q
    float mq = -1e30f, lq = 0.0f;      // per-lane softmax state (query lr)

    const size_t mbase = ((size_t)b * Sq + (q0 + lr)) * Sq;

    stage(0, 0);
    for (int kb = 0; kb < NBk; ++kb) {
        const int cur = kb & 1;
        __syncthreads();                       // prev compute done before overwrite
        if (kb + 1 < NBk) stage(kb + 1, cur ^ 1);
#if ATTN_ASYNC
        if (kb + 1 < NBk) __builtin_amdgcn_s_wait_asynccnt(4);  // cur block landed
        else              __builtin_amdgcn_s_wait_asynccnt(0);
#endif
        __syncthreads();                       // cur buffer visible to all waves

        // ---- S^T = K x Q^T : 4 key tiles x 2 hd steps ----
        v8f sc[4] = {};
#pragma unroll
        for (int t = 0; t < 4; ++t) {
#pragma unroll
            for (int s2 = 0; s2 < 2; ++s2) {
                bf16x8 alo = *(const bf16x8*)&Ks[cur][t * 16 + lr][s2 * 32 + kh8];
                bf16x8 ahi = *(const bf16x8*)&Ks[cur][t * 16 + lr][s2 * 32 + kh8 + 16];
                v16bf ka = make_v16(alo, ahi);
                sc[t] = __builtin_amdgcn_wmma_f32_16x16x32_bf16(
                    false, ka, false, qb[s2], (short)0, sc[t], false, false);
            }
        }

        // ---- mask: per lane keys are contiguous -> two int4 loads per tile ----
#pragma unroll
        for (int t = 0; t < 4; ++t) {
            const int4* mp = (const int4*)&mask[mbase + kb * KBk + t * 16 + kh8];
            int4 ma = mp[0], mb2 = mp[1];
            int mv[8] = {ma.x, ma.y, ma.z, ma.w, mb2.x, mb2.y, mb2.z, mb2.w};
#pragma unroll
            for (int r = 0; r < 8; ++r)
                sc[t][r] = (mv[r] == 0) ? -1e9f : sc[t][r];
        }

        // ---- online softmax: in-lane reduce + one shfl_xor(16) ----
        float rm = -1e30f;
#pragma unroll
        for (int t = 0; t < 4; ++t)
#pragma unroll
            for (int r = 0; r < 8; ++r) rm = fmaxf(rm, sc[t][r]);
        rm = fmaxf(rm, __shfl_xor(rm, 16, 32));
        const float mnew = fmaxf(mq, rm);
        const float corr = __expf(mq - mnew);
        mq = mnew;
        float rs = 0.0f;
#pragma unroll
        for (int t = 0; t < 4; ++t)
#pragma unroll
            for (int r = 0; r < 8; ++r) {
                float p = __expf(sc[t][r] - mnew);
                sc[t][r] = p;
                rs += p;
            }
        rs += __shfl_xor(rs, 16, 32);
        lq = lq * corr + rs;

        // rescale accumulator rows (row r holds query r + 8*half)
#pragma unroll
        for (int r = 0; r < 8; ++r) {
            const float cr = __shfl(corr, r + 8 * half, 32);
#pragma unroll
            for (int u = 0; u < 4; ++u) acc[u][r] *= cr;
        }

        // ---- pack P: C-layout of S^T is exactly the A-layout of P ----
        v16bf pa[2];
#pragma unroll
        for (int s2 = 0; s2 < 2; ++s2) {
            V16U u;
#pragma unroll
            for (int r = 0; r < 8; ++r) {
                u.h[0][r] = (bf16)sc[2 * s2 + 0][r];
                u.h[1][r] = (bf16)sc[2 * s2 + 1][r];
            }
            pa[s2] = u.v;
        }

        // ---- O += P x V : 2 k-steps x 4 hd tiles, V in natural layout ----
#pragma unroll
        for (int s2 = 0; s2 < 2; ++s2) {
#pragma unroll
            for (int u = 0; u < 4; ++u) {
                bf16x8 blo, bhi;
#pragma unroll
                for (int j = 0; j < 8; ++j) {
                    blo[j] = Vs[cur][s2 * 32 + kh8 + j][u * 16 + lr];
                    bhi[j] = Vs[cur][s2 * 32 + kh8 + 16 + j][u * 16 + lr];
                }
                v16bf vb = make_v16(blo, bhi);
                acc[u] = __builtin_amdgcn_wmma_f32_16x16x32_bf16(
                    false, pa[s2], false, vb, (short)0, acc[u], false, false);
            }
        }
    }

    // ---- epilogue: normalize, store bf16 [B,S,D] ----
    const float inv = 1.0f / lq;
#pragma unroll
    for (int r = 0; r < 8; ++r) {
        const float ir = __shfl(inv, r + 8 * half, 32);
        const int sr   = q0 + r + 8 * half;
#pragma unroll
        for (int u = 0; u < 4; ++u)
            Outb[((size_t)b * Sq + sr) * Dm + h * HDm + u * 16 + lr] =
                (bf16)(acc[u][r] * ir);
    }
}

// ---------------------------------------------------------------------------
extern "C" void kernel_launch(void* const* d_in, const int* in_sizes, int n_in,
                              void* d_out, int out_size, void* d_ws, size_t ws_size,
                              hipStream_t stream) {
    (void)in_sizes; (void)n_in; (void)out_size; (void)ws_size;

    const float* query = (const float*)d_in[0];
    const float* key   = (const float*)d_in[1];
    const float* value = (const float*)d_in[2];
    const int*   mask  = (const int*)d_in[3];
    const float* Wq = (const float*)d_in[4];
    const float* bq = (const float*)d_in[5];
    const float* Wk = (const float*)d_in[6];
    const float* bk = (const float*)d_in[7];
    const float* Wv = (const float*)d_in[8];
    const float* bv = (const float*)d_in[9];
    const float* Wo = (const float*)d_in[10];
    const float* bo = (const float*)d_in[11];

    const size_t NE = (size_t)Bc * Sq * Dm;
    bf16* Qbf  = (bf16*)d_ws;        // [B,H,S,HD]
    bf16* Kbf  = Qbf + NE;
    bf16* Vbf  = Kbf + NE;
    bf16* Attn = Vbf + NE;           // [B,S,D]

    dim3 pgrid(Bc * Sq / 32, Dm / 64);
    dim3 pblk(256);

    proj_kernel<false, true><<<pgrid, pblk, 0, stream>>>(query, Wq, bq, Qbf);
    proj_kernel<false, true><<<pgrid, pblk, 0, stream>>>(key,   Wk, bk, Kbf);
    proj_kernel<false, true><<<pgrid, pblk, 0, stream>>>(value, Wv, bv, Vbf);

    dim3 agrid(Sq / 128, Bc * Hn);
    attn_kernel<<<agrid, dim3(256), 0, stream>>>(Qbf, Kbf, Vbf, mask, Attn);

    proj_kernel<true, false><<<pgrid, pblk, 0, stream>>>(Attn, Wo, bo, (float*)d_out);
}
